// Embedding_11433202942435
// MI455X (gfx1250) — matrix operand order repaired and checked by reading, not compiled, not run on hardware
//
#include <hip/hip_runtime.h>
#include <cstdint>

// ---------------------------------------------------------------------------
// MI455X / gfx1250 implementation of the 3-layer edge-message GNN.
// bf16 WMMA (f32 accumulate) for every GEMM, fused [Wmsg;Wedge] stacked-K
// message GEMM, TDM (tensor_load_to_lds) weight staging, fragment-swizzled
// LDS layout, f32 scatter atomics. wave32: one wave = one 16-row tile.
// ---------------------------------------------------------------------------

typedef __bf16 bf16;
typedef __attribute__((ext_vector_type(16))) __bf16 v16bf;
typedef __attribute__((ext_vector_type(8)))  float  v8f;
typedef __attribute__((ext_vector_type(4)))  unsigned int u32x4;
typedef __attribute__((ext_vector_type(8)))  int i32x8;
typedef __attribute__((ext_vector_type(4)))  int i32x4;

#define WMMA_BF16(A,B,C) \
  __builtin_amdgcn_wmma_f32_16x16x32_bf16(false,(A),false,(B),(short)0,(C),false,false)

#if defined(__HIP_DEVICE_COMPILE__) && __has_builtin(__builtin_amdgcn_tensor_load_to_lds)
#define HAVE_TDM 1
#else
#define HAVE_TDM 0
#endif

// Problem constants (from reference)
constexpr int NNODES = 50000;
constexpr int NEDGES = 800000;
// Output tuple offsets (floats): h, node_attr, src, dst, ea, basis, batch
constexpr size_t OH     = 0;
constexpr size_t ONA    = 1600000;
constexpr size_t OSRC   = 1650000;
constexpr size_t ODST   = 2450000;
constexpr size_t OEA    = 3250000;
constexpr size_t OB     = 13650000;
constexpr size_t OBATCH = 21650000;

// Workspace layout (bytes)
constexpr size_t WS_HCUR  = 0;          // bf16 [N][160]   (16,000,000 B)
constexpr size_t WS_HNEXT = 16000000;   // f32  [N][160]   (32,000,000 B)
constexpr size_t WS_EABF  = 48000000;   // bf16 [E][16]    (25,600,000 B)
constexpr size_t WS_WTS   = 73600000;   // packed bf16 weights

static_assert(NNODES % 16 == 0, "node tiles");
static_assert(NEDGES % 16 == 0, "edge tiles");

// Branch-free fast gelu (tanh form): keeps EXEC uniform around WMMA regions.
__device__ __forceinline__ float gelu_f(float v) {
  const float k0 = 0.7978845608028654f; // sqrt(2/pi)
  float u = k0 * (v + 0.044715f * v * v * v);
  float a = fminf(fabsf(u), 8.0f);
  float t = __expf(-2.0f * a);                       // v_exp_f32
  float th = (1.0f - t) * __builtin_amdgcn_rcpf(1.0f + t);
  th = copysignf(th, u);
  return 0.5f * v * (1.0f + th);
}

__device__ __forceinline__ v8f zero8() {
  v8f z = {0.f,0.f,0.f,0.f,0.f,0.f,0.f,0.f};
  return z;
}

// ---- TDM: async DMA of a flat bf16 array (global -> LDS), D# per ISA §8 ----
// 1-row tensor: tensor_dim0 = tile_dim0 = nelem (< 65536), data_size = 2B.
__device__ __forceinline__ void tdm_load_b16_1d(const void* gsrc, void* lds_dst,
                                                unsigned nelem) {
#if HAVE_TDM
  unsigned long long ga = (unsigned long long)gsrc;
  u32x4 g0;
  g0[0] = 1u;                                           // count=1 (valid), user
  g0[1] = (unsigned)(unsigned long long)lds_dst;        // lds_addr (bytes)
  g0[2] = (unsigned)ga;                                 // global_addr[31:0]
  g0[3] = (unsigned)((ga >> 32) & 0x01FFFFFFu) | 0x80000000u; // addr[56:32]|type=2
  i32x8 g1;
  g1[0] = (int)(1u << 16);                              // data_size=1 (2 bytes)
  g1[1] = (int)((nelem & 0xFFFFu) << 16);               // tensor_dim0[15:0] @63:48
  g1[2] = (int)(((nelem >> 16) & 0xFFFFu) | (1u << 16));// dim0[31:16] | dim1 lo=1
  g1[3] = (int)((nelem & 0xFFFFu) << 16);               // dim1 hi=0 | tile_dim0
  g1[4] = 1;                                            // tile_dim1=1, tile_dim2=0
  g1[5] = (int)nelem;                                   // tensor_dim0_stride lo
  g1[6] = 0;
  g1[7] = 0;
  i32x4 z4 = {0, 0, 0, 0};
#if __clang_major__ >= 23
  i32x8 z8 = {0, 0, 0, 0, 0, 0, 0, 0};
  __builtin_amdgcn_tensor_load_to_lds(g0, g1, z4, z4, z8, 0);
#else
  __builtin_amdgcn_tensor_load_to_lds(g0, g1, z4, z4, 0);
#endif
#else
  (void)gsrc; (void)lds_dst; (void)nelem;
#endif
}

__device__ __forceinline__ void tdm_wait() {
#if defined(__HIP_DEVICE_COMPILE__) && __has_builtin(__builtin_amdgcn_s_wait_tensorcnt)
  __builtin_amdgcn_s_wait_tensorcnt(0);
#endif
}

// ---- WMMA fragment helpers (wave32 layouts per CDNA5 ISA 7.12.2) ----------
// B fragment, pre-swizzled: element i of lane l = B[kt*32+8*(l>=16)+i+(i&8)][nt*16+(l&15)]
// stored contiguously: 512 bf16 per (kt,nt) fragment -> one 32B load per lane.
__device__ __forceinline__ v16bf load_frag_sw(const bf16* s, int NT, int kt, int nt, int lane) {
  return *(const v16bf*)(s + (((size_t)(kt*NT + nt)) << 9) + (lane << 4));
}
// A fragment from row-major [16][stride] bf16 in LDS: element e -> k = k0 + e + (e&8) + 8*(l>=16)
__device__ __forceinline__ v16bf load_A_rm(const bf16* s, int stride, int lane, int k0) {
  int m = lane & 15, hf = lane >> 4;
  v16bf a;
#pragma unroll
  for (int e = 0; e < 16; ++e) {
    int k = k0 + e + (e & 8) + (hf << 3);
    a[e] = s[m * stride + k];
  }
  return a;
}
// swizzled pack decode: flat index t -> (row, col) of the source matrix
__device__ __forceinline__ void sw_decode(int t, int Ncol, int& r, int& c) {
  int frag = t >> 9, rem = t & 511, lane = rem >> 4, i = rem & 15;
  int NT = Ncol >> 4;
  int kt = frag / NT, nt = frag - kt * NT;
  r = kt * 32 + ((lane >> 4) << 3) + i + (i & 8);
  c = (nt << 4) + (lane & 15);
}

// ---------------------------------------------------------------------------
// Weight packing: fold 1/sqrt scales, convert to bf16, store fragment-swizzled.
// ---------------------------------------------------------------------------
__global__ void pack_weights_kernel(
    const float* __restrict__ embW1, const float* __restrict__ embW2,
    const float* Wm0, const float* We0, const float* Ws0, const float* f20,
    const float* Wm1, const float* We1, const float* Ws1, const float* f21,
    const float* Wm2, const float* We2, const float* Ws2, const float* f22,
    char* __restrict__ wts)
{
  const float* Wm[3] = {Wm0, Wm1, Wm2};
  const float* We[3] = {We0, We1, We2};
  const float* Ws[3] = {Ws0, Ws1, Ws2};
  const float* f2[3] = {f20, f21, f22};
  const int DIN[3]   = {32, 144, 144};
  const int DOUT[3]  = {144, 144, 32};
  const int KMSG[3]  = {64, 160, 160};
  const int KSELF[3] = {32, 160, 160};
  const float invsd[3] = {0.17677669529663687f, 0.0833333333333333f, 0.0833333333333333f};
  const float inv32 = 0.17677669529663687f;  // 1/sqrt(MUL)
  const float inv13 = 0.2773500981126146f;   // 1/sqrt(EA+SHD)

  int tid = blockIdx.x * blockDim.x + threadIdx.x;
  int stp = gridDim.x * blockDim.x;

  bf16* w1sw = (bf16*)(wts);
  for (int t = tid; t < 64 * 32; t += stp) {
    int r, c; sw_decode(t, 32, r, c);
    w1sw[t] = (bf16)(embW1[r * 32 + c] * 0.125f);         // 1/sqrt(INPUT_DIM)
  }
  bf16* w2sw = (bf16*)(wts + 4096);
  for (int t = tid; t < 32 * 32; t += stp) {
    int r, c; sw_decode(t, 32, r, c);
    w2sw[t] = (bf16)(embW2[r * 32 + c] * inv32);
  }
  for (int i = 0; i < 3; ++i) {
    char* lw = wts + 8192 + i * 131072;
    int din = DIN[i], dout = DOUT[i];
    bf16* fsw = (bf16*)lw;
    for (int t = tid; t < 32 * din; t += stp) {
      int r, c; sw_decode(t, din, r, c);
      fsw[t] = (bf16)(f2[i][r * din + c] * inv32);
    }
    bf16* bm = (bf16*)(lw + 16384);
    for (int t = tid; t < KMSG[i] * dout; t += stp) {
      int r, c; sw_decode(t, dout, r, c);
      float v = 0.f;
      if (r < din)            v = Wm[i][r * dout + c] * invsd[i];
      else if (r < din + 13)  v = We[i][(r - din) * dout + c] * inv13;
      bm[t] = (bf16)v;
    }
    bf16* bs = (bf16*)(lw + 65536);
    for (int t = tid; t < KSELF[i] * dout; t += stp) {
      int r, c; sw_decode(t, dout, r, c);
      float v = (r < din) ? Ws[i][r * dout + c] * invsd[i] : 0.f;
      bs[t] = (bf16)v;
    }
  }
}

__global__ void zero_u32_kernel(uint32_t* p, long n) {
  long i = (long)blockIdx.x * blockDim.x + threadIdx.x;
  long st = (long)gridDim.x * blockDim.x;
  for (; i < n; i += st) p[i] = 0u;
}

// ---------------------------------------------------------------------------
// Edge geometry: sh2, ea (f32 out + bf16 padded-to-16 copy), radial basis,
// src/dst as float outputs.
// ---------------------------------------------------------------------------
__global__ void edge_geom_kernel(const float* __restrict__ pos,
                                 const int* __restrict__ eidx,
                                 const float* __restrict__ eattr,
                                 float* __restrict__ out,
                                 bf16* __restrict__ eabf)
{
  int e = blockIdx.x * 256 + threadIdx.x;
  if (e >= NEDGES) return;
  int s = eidx[e], d = eidx[NEDGES + e];
  out[OSRC + e] = (float)s;
  out[ODST + e] = (float)d;
  float ex = pos[3 * s + 0] - pos[3 * d + 0];
  float ey = pos[3 * s + 1] - pos[3 * d + 1];
  float ez = pos[3 * s + 2] - pos[3 * d + 2];
  float r = sqrtf(ex * ex + ey * ey + ez * ez + 1e-12f);
  float inv = 1.0f / r;
  float nx = ex * inv, ny = ey * inv, nz = ez * inv;
  const float s3 = 1.7320508075688772f, s5 = 2.23606797749979f, s15 = 3.872983346207417f;
  float sh[9];
  sh[0] = 1.f;        sh[1] = s3 * nx;    sh[2] = s3 * ny;    sh[3] = s3 * nz;
  sh[4] = s15 * nx * nz;  sh[5] = s15 * nx * ny;
  sh[6] = s5 * (ny * ny - 0.5f * (nx * nx + nz * nz));
  sh[7] = s15 * ny * nz;  sh[8] = 0.5f * s15 * (nz * nz - nx * nx);
  size_t ob = OEA + (size_t)e * 13;
  size_t eb = (size_t)e * 16;
#pragma unroll
  for (int c = 0; c < 13; ++c) {
    float v = (c < 4) ? eattr[(size_t)e * 4 + c] : sh[c - 4];
    out[ob + c] = v;
    eabf[eb + c] = (bf16)v;
  }
  eabf[eb + 13] = (bf16)0.f; eabf[eb + 14] = (bf16)0.f; eabf[eb + 15] = (bf16)0.f;
  const float step = 2.0f / 11.0f;
  size_t bb = OB + (size_t)e * 10;
#pragma unroll
  for (int j = 0; j < 10; ++j) {
    float ctr = step * (j + 1);
    float dd = (r - ctr) / step;
    float v = (dd > -1.f && dd < 1.f) ? cosf(1.5707963267948966f * dd) * 3.1622776601683795f : 0.f;
    out[bb + j] = v;
  }
}

__global__ void copy_nodes_kernel(const float* __restrict__ na,
                                  const int* __restrict__ batch,
                                  float* __restrict__ out)
{
  int i = blockIdx.x * 256 + threadIdx.x;
  if (i >= NNODES) return;
  out[ONA + i] = na[i];
  out[OBATCH + i] = (float)batch[i];
}

// ---------------------------------------------------------------------------
// Embedding: h0 = gelu(x @ W1') @ W2'  (16-node tile per wave, 6 WMMAs)
// ---------------------------------------------------------------------------
__global__ __launch_bounds__(128) void embed_kernel(
    const float* __restrict__ x, bf16* __restrict__ hcur,
    const bf16* __restrict__ gW1, const bf16* __restrict__ gW2, int ntiles)
{
  constexpr int WAVES = 4;
  extern __shared__ char smem[];
  bf16* sW1  = (bf16*)smem;          // 64x32 swizzled (2048)
  bf16* sW2  = sW1 + 2048;           // 32x32 swizzled (1024)
  bf16* xA   = sW2 + 1024;           // per-wave 16x64 row-major
  bf16* midA = xA + WAVES * 1024;    // per-wave 16x32 row-major

  int tid = threadIdx.x;
  int l = tid & 31, wv = tid >> 5;
#if HAVE_TDM
  if (wv == 0) {                      // gW1,gW2 contiguous in ws: one DMA
    tdm_load_b16_1d(gW1, sW1, 3072);
    tdm_wait();
  }
#else
  for (int t = tid; t < 1024; t += 128) ((uint32_t*)sW1)[t] = ((const uint32_t*)gW1)[t];
  for (int t = tid; t < 512;  t += 128) ((uint32_t*)sW2)[t] = ((const uint32_t*)gW2)[t];
#endif
  __syncthreads();

  int m = l & 15, hf = l >> 4;
  int tile = blockIdx.x * WAVES + wv;
  if (tile >= ntiles) return;
  int n0 = tile << 4;

  bf16* xa = xA + (wv << 10);
  for (int t = l; t < 1024; t += 32)
    xa[t] = (bf16)x[(size_t)(n0 + (t >> 6)) * 64 + (t & 63)];

  bf16* mid = midA + (wv << 9);
#pragma unroll
  for (int nt = 0; nt < 2; ++nt) {
    v8f a8 = zero8();
#pragma unroll
    for (int kt = 0; kt < 2; ++kt) {
      v16bf a = load_A_rm(xa, 64, l, kt * 32);
      v16bf b = load_frag_sw(sW1, 2, kt, nt, l);
      a8 = WMMA_BF16(a, b, a8);
    }
#pragma unroll
    for (int r = 0; r < 8; ++r)
      mid[(r + (hf << 3)) * 32 + (nt << 4) + m] = (bf16)gelu_f(a8[r]);
  }
  v16bf a2 = load_A_rm(mid, 32, l, 0);
#pragma unroll
  for (int nt = 0; nt < 2; ++nt) {
    v8f a8 = zero8();
    v16bf b = load_frag_sw(sW2, 2, 0, nt, l);
    a8 = WMMA_BF16(a2, b, a8);
#pragma unroll
    for (int r = 0; r < 8; ++r)
      hcur[(size_t)(n0 + r + (hf << 3)) * 160 + (nt << 4) + m] = (bf16)a8[r];
  }
}

// ---------------------------------------------------------------------------
// Message kernel: per 16-edge tile (one wave):
//  hidden = gelu(basis/sqrt(10) @ fcW1)      (VALU, 16x32)
//  w      = hidden @ fcW2'                   (WMMA, din/16 frags)
//  A      = [h[src]*w , ea]  16 x KMSG       (bf16, gathered)
//  m      = A @ [Wmsg';Wedge']               (WMMA chain, KMSG/32 x dout/16)
//  scatter: atomicAdd(hnext[dst], m * 0.25)  (1/sqrt(NNEI) folded)
// ---------------------------------------------------------------------------
template<int DIN, int DOUT, int KMSG>
__global__ __launch_bounds__(128) void message_kernel(
    const bf16* __restrict__ hcur, float* __restrict__ hnext,
    const bf16* __restrict__ eabf, const float* __restrict__ basisbuf,
    const int* __restrict__ srcIdx, const int* __restrict__ dstIdx,
    const bf16* __restrict__ gBmsg, const bf16* __restrict__ gFc,
    const float* __restrict__ gFcW1)
{
  constexpr int NTO = DOUT / 16, NTD = DIN / 16, NKT = KMSG / 32, WAVES = 4;
  extern __shared__ char smem[];
  bf16* sB    = (bf16*)smem;                   // KMSG x DOUT (swizzled)
  bf16* sFc   = sB + KMSG * DOUT;              // 32 x DIN (swizzled)
  float* sW1  = (float*)(sFc + 32 * DIN);      // fcW1 10x32 f32
  float* wbuf = sW1 + 320;                     // per-wave 16 x DIN f32
  bf16* hidA  = (bf16*)(wbuf + WAVES * 16 * DIN); // per-wave 16x32
  int* sidx   = (int*)(hidA + WAVES * 512);    // per-wave src[16],dst[16]

  int tid = threadIdx.x;
  int l = tid & 31, wv = tid >> 5;
#if HAVE_TDM
  if (wv == 0) {                 // TDM DMA of both weight blocks, then wait
    tdm_load_b16_1d(gBmsg, sB, (unsigned)(KMSG * DOUT));
    tdm_load_b16_1d(gFc, sFc, (unsigned)(32 * DIN));
    tdm_wait();
  }
#else
  for (int t = tid; t < KMSG * DOUT / 2; t += 128) ((uint32_t*)sB)[t]  = ((const uint32_t*)gBmsg)[t];
  for (int t = tid; t < 32 * DIN / 2;    t += 128) ((uint32_t*)sFc)[t] = ((const uint32_t*)gFc)[t];
#endif
  for (int t = tid; t < 320; t += 128) sW1[t] = gFcW1[t];
  __syncthreads();

  int m = l & 15, hf = l >> 4;
  int tile = blockIdx.x * WAVES + wv;          // grid sized so all tiles valid
  int e0 = tile << 4;

  if (l < 16) {
    sidx[(wv << 5) + l]      = srcIdx[e0 + l];
    sidx[(wv << 5) + 16 + l] = dstIdx[e0 + l];
  }

  // radial MLP hidden (each lane: row m, 16 columns of its half)
  float bbv[10];
#pragma unroll
  for (int j = 0; j < 10; ++j) bbv[j] = basisbuf[(size_t)(e0 + m) * 10 + j];
  bf16* ha = hidA + (wv << 9);
#pragma unroll
  for (int nn = 0; nn < 16; ++nn) {
    int n = (hf << 4) + nn;
    float a = 0.f;
#pragma unroll
    for (int j = 0; j < 10; ++j) a += bbv[j] * sW1[j * 32 + n];
    ha[m * 32 + n] = (bf16)gelu_f(a * 0.31622776601683794f);  // 1/sqrt(NB)
  }

  // w = hidden @ fcW2'
  float* wb = wbuf + wv * 16 * DIN;
  v16bf ah = load_A_rm(ha, 32, l, 0);
#pragma unroll
  for (int nt = 0; nt < NTD; ++nt) {
    v16bf bfr = load_frag_sw(sFc, NTD, 0, nt, l);
    v8f a8 = zero8();
    a8 = WMMA_BF16(ah, bfr, a8);
#pragma unroll
    for (int r = 0; r < 8; ++r)
      wb[(r + (hf << 3)) * DIN + (nt << 4) + m] = a8[r];
  }

  // fused message GEMM
  v8f acc[NTO];
#pragma unroll
  for (int nt = 0; nt < NTO; ++nt) acc[nt] = zero8();
  int sE = sidx[(wv << 5) + m];
  const bf16* hrow = hcur + (size_t)sE * 160;
  __builtin_prefetch(hrow, 0, 1);   // global_prefetch of the gathered h row
  size_t eb = (size_t)(e0 + m) << 4;
#pragma unroll
  for (int kt = 0; kt < NKT; ++kt) {
    v16bf a;
#pragma unroll
    for (int p = 0; p < 8; ++p) {
      int e = p << 1;
      int k = kt * 32 + e + (e & 8) + (hf << 3);
      if (k < DIN) {
        a[e]     = (bf16)((float)hrow[k]     * wb[m * DIN + k]);
        a[e + 1] = (bf16)((float)hrow[k + 1] * wb[m * DIN + k + 1]);
      } else if (k < DIN + 16) {
        a[e]     = eabf[eb + (k - DIN)];
        a[e + 1] = eabf[eb + (k + 1 - DIN)];
      } else {
        a[e] = (bf16)0.f; a[e + 1] = (bf16)0.f;
      }
    }
#pragma unroll
    for (int nt = 0; nt < NTO; ++nt) {
      v16bf bfr = load_frag_sw(sB, NTO, kt, nt, l);
      acc[nt] = WMMA_BF16(a, bfr, acc[nt]);
    }
  }

  // scatter-add (segment_sum / sqrt(16))
#pragma unroll
  for (int nt = 0; nt < NTO; ++nt) {
#pragma unroll
    for (int r = 0; r < 8; ++r) {
      int mm = r + (hf << 3);
      int col = (nt << 4) + m;
      int dn = sidx[(wv << 5) + 16 + mm];
      atomicAdd(&hnext[(size_t)dn * 160 + col], acc[nt][r] * 0.25f);
    }
  }
}

// ---------------------------------------------------------------------------
// Node update: h = agg + h @ Wself' + node_attr @ Wattr  (+gelu)
// ---------------------------------------------------------------------------
template<int DIN, int DOUT, int KSELF, bool DO_GELU, bool WRITE_OUT>
__global__ __launch_bounds__(128) void node_update_kernel(
    bf16* __restrict__ hcur, const float* __restrict__ hnext,
    const float* __restrict__ node_attr, const bf16* __restrict__ gBself,
    const float* __restrict__ gWattr, float* __restrict__ outH, int ntiles)
{
  constexpr int NTO = DOUT / 16, NKT = KSELF / 32, WAVES = 4;
  extern __shared__ char smem[];
  bf16* sB   = (bf16*)smem;                 // KSELF x DOUT (swizzled)
  float* sWa = (float*)(sB + KSELF * DOUT); // Wattr row
  int tid = threadIdx.x;
  int l = tid & 31, wv = tid >> 5;
#if HAVE_TDM
  if (wv == 0) {
    tdm_load_b16_1d(gBself, sB, (unsigned)(KSELF * DOUT));
    tdm_wait();
  }
#else
  for (int t = tid; t < KSELF * DOUT / 2; t += 128) ((uint32_t*)sB)[t] = ((const uint32_t*)gBself)[t];
#endif
  for (int t = tid; t < DOUT; t += 128) sWa[t] = gWattr[t];
  __syncthreads();

  int m = l & 15, hf = l >> 4;
  int tile = blockIdx.x * WAVES + wv;
  if (tile >= ntiles) return;
  int n0 = tile << 4;

  const bf16* hrow = hcur + (size_t)(n0 + m) * 160;
  v8f acc[NTO];
#pragma unroll
  for (int nt = 0; nt < NTO; ++nt) acc[nt] = zero8();
#pragma unroll
  for (int kt = 0; kt < NKT; ++kt) {
    v16bf a;
#pragma unroll
    for (int p = 0; p < 8; ++p) {
      int e = p << 1;
      int k = kt * 32 + e + (e & 8) + (hf << 3);
      if (k < DIN) { a[e] = hrow[k]; a[e + 1] = hrow[k + 1]; }
      else         { a[e] = (bf16)0.f; a[e + 1] = (bf16)0.f; }
    }
#pragma unroll
    for (int nt = 0; nt < NTO; ++nt) {
      v16bf bfr = load_frag_sw(sB, NTO, kt, nt, l);
      acc[nt] = WMMA_BF16(a, bfr, acc[nt]);
    }
  }
#pragma unroll
  for (int nt = 0; nt < NTO; ++nt) {
#pragma unroll
    for (int r = 0; r < 8; ++r) {
      int node = n0 + r + (hf << 3);
      int col = (nt << 4) + m;
      float v = acc[nt][r] + hnext[(size_t)node * 160 + col] + node_attr[node] * sWa[col];
      if (DO_GELU) v = gelu_f(v);
      if (WRITE_OUT) outH[(size_t)node * 32 + col] = v;
      else           hcur[(size_t)node * 160 + col] = (bf16)v;
    }
  }
}

// ---------------------------------------------------------------------------
constexpr size_t msg_smem(int DIN, int DOUT, int KMSG) {
  return (size_t)KMSG * DOUT * 2 + 32 * DIN * 2 + 320 * 4
       + 4 * 16 * (size_t)DIN * 4 + 4 * 512 * 2 + 4 * 32 * 4;
}
constexpr size_t nod_smem(int KSELF, int DOUT) {
  return (size_t)KSELF * DOUT * 2 + (size_t)DOUT * 4;
}

extern "C" void kernel_launch(void* const* d_in, const int* in_sizes, int n_in,
                              void* d_out, int out_size, void* d_ws, size_t ws_size,
                              hipStream_t stream) {
  (void)in_sizes; (void)n_in; (void)out_size; (void)ws_size;
  // Assumed pytree-leaf order: pos, x, node_attr, edge_index, edge_attr, batch,
  // then params with sorted keys: embed_W1, embed_W2,
  // layers[i]: Wattr, Wedge, Wmsg, Wself, fcW1, fcW2.  (int inputs as int32)
  const float* pos       = (const float*)d_in[0];
  const float* x         = (const float*)d_in[1];
  const float* node_attr = (const float*)d_in[2];
  const int*   eidx      = (const int*)d_in[3];
  const float* eattr     = (const float*)d_in[4];
  const int*   batch     = (const int*)d_in[5];
  const float* embW1     = (const float*)d_in[6];
  const float* embW2     = (const float*)d_in[7];
  const float *Wattr[3], *Wedge[3], *Wmsg[3], *Wself[3], *fcW1[3], *fcW2[3];
  for (int i = 0; i < 3; ++i) {
    int b = 8 + 6 * i;
    Wattr[i] = (const float*)d_in[b + 0];
    Wedge[i] = (const float*)d_in[b + 1];
    Wmsg[i]  = (const float*)d_in[b + 2];
    Wself[i] = (const float*)d_in[b + 3];
    fcW1[i]  = (const float*)d_in[b + 4];
    fcW2[i]  = (const float*)d_in[b + 5];
  }

  char* ws = (char*)d_ws;
  bf16*  hcur  = (bf16*)(ws + WS_HCUR);
  float* hnext = (float*)(ws + WS_HNEXT);
  bf16*  eabf  = (bf16*)(ws + WS_EABF);
  char*  wts   = ws + WS_WTS;
  bf16* W1sw = (bf16*)(wts);
  bf16* W2sw = (bf16*)(wts + 4096);
  bf16 *fcW2sw[3], *Bmsg[3], *Bself[3];
  for (int i = 0; i < 3; ++i) {
    char* lw = wts + 8192 + (size_t)i * 131072;
    fcW2sw[i] = (bf16*)lw;
    Bmsg[i]   = (bf16*)(lw + 16384);
    Bself[i]  = (bf16*)(lw + 65536);
  }
  float* out = (float*)d_out;

  pack_weights_kernel<<<512, 256, 0, stream>>>(embW1, embW2,
      Wmsg[0], Wedge[0], Wself[0], fcW2[0],
      Wmsg[1], Wedge[1], Wself[1], fcW2[1],
      Wmsg[2], Wedge[2], Wself[2], fcW2[2], wts);

  zero_u32_kernel<<<2048, 256, 0, stream>>>((uint32_t*)hcur, 4000000L);   // hcur = 0
  edge_geom_kernel<<<NEDGES / 256, 256, 0, stream>>>(pos, eidx, eattr, out, eabf);
  copy_nodes_kernel<<<(NNODES + 255) / 256, 256, 0, stream>>>(node_attr, batch, out);
  embed_kernel<<<782, 128, 18432, stream>>>(x, hcur, W1sw, W2sw, 3125);

  const int MSG_BLOCKS = NEDGES / 16 / 4;   // 12500
  // layer 0: din=32, dout=144, KMSG=64(32+13 pad), KSELF=32
  zero_u32_kernel<<<2048, 256, 0, stream>>>((uint32_t*)hnext, 8000000L);
  message_kernel<32, 144, 64><<<MSG_BLOCKS, 128, msg_smem(32, 144, 64), stream>>>(
      hcur, hnext, eabf, out + OB, eidx, eidx + NEDGES, Bmsg[0], fcW2sw[0], fcW1[0]);
  node_update_kernel<32, 144, 32, true, false><<<782, 128, nod_smem(32, 144), stream>>>(
      hcur, hnext, node_attr, Bself[0], Wattr[0], out, 3125);
  // layer 1: 144 -> 144
  zero_u32_kernel<<<2048, 256, 0, stream>>>((uint32_t*)hnext, 8000000L);
  message_kernel<144, 144, 160><<<MSG_BLOCKS, 128, msg_smem(144, 144, 160), stream>>>(
      hcur, hnext, eabf, out + OB, eidx, eidx + NEDGES, Bmsg[1], fcW2sw[1], fcW1[1]);
  node_update_kernel<144, 144, 160, true, false><<<782, 128, nod_smem(160, 144), stream>>>(
      hcur, hnext, node_attr, Bself[1], Wattr[1], out, 3125);
  // layer 2: 144 -> 32 (final, no gelu, write f32 h to d_out)
  zero_u32_kernel<<<2048, 256, 0, stream>>>((uint32_t*)hnext, 8000000L);
  message_kernel<144, 32, 160><<<MSG_BLOCKS, 128, msg_smem(144, 32, 160), stream>>>(
      hcur, hnext, eabf, out + OB, eidx, eidx + NEDGES, Bmsg[2], fcW2sw[2], fcW1[2]);
  node_update_kernel<144, 32, 160, false, true><<<782, 128, nod_smem(160, 32), stream>>>(
      hcur, hnext, node_attr, Bself[2], Wattr[2], out + OH, 3125);
}
